// MHAtt_59906203845364
// MI455X (gfx1250) — matrix operand order repaired and checked by reading, not compiled, not run on hardware
//
#include <hip/hip_runtime.h>
#include <hip/hip_bf16.h>

#define HID 1024
#define NHEAD 16
#define DHEAD 64
#define SEQ 2048
#define BATCH 2

typedef __attribute__((ext_vector_type(16))) __bf16        v16bf;
typedef __attribute__((ext_vector_type(8)))  float         v8f;
typedef __attribute__((ext_vector_type(4)))  unsigned int  u32x4;
typedef __attribute__((ext_vector_type(2)))  unsigned int  u32x2;
typedef __attribute__((ext_vector_type(4)))  float         f32x4;

union Frag16 { u32x4 q[2]; v16bf v; };

static __device__ inline unsigned short f2bf(float x) {
    __bf16 h = (__bf16)x;
    return __builtin_bit_cast(unsigned short, h);
}

// Pack two fp32 -> packed bf16 pair
static __device__ inline unsigned int pk2(float lo, float hi) {
#if __has_builtin(__builtin_amdgcn_cvt_pk_bf16_f32)
    auto r = __builtin_amdgcn_cvt_pk_bf16_f32(lo, hi);
    return __builtin_bit_cast(unsigned int, r);
#else
    return (unsigned int)f2bf(lo) | ((unsigned int)f2bf(hi) << 16);
#endif
}

// v_wmma_f32_16x16x32_bf16 wrapper (8-arg form)
static __device__ inline v8f wmma_bf16(v16bf a, v16bf b, v8f c) {
    return __builtin_amdgcn_wmma_f32_16x16x32_bf16(
        /*neg_a=*/false, a, /*neg_b=*/false, b,
        /*c_mod=*/(short)0, c, /*reuse_a=*/false, /*reuse_b=*/false);
}

// A fragment (16x32, bf16) from row-major bf16 matrix, ld elements.
static __device__ inline v16bf load_afrag_bf(const unsigned short* base, int ld,
                                             int m0, int kb, int lane) {
    int m = lane & 15, hh = lane >> 4;
    const unsigned short* p = base + (size_t)(m0 + m) * ld + kb + 16 * hh;
    Frag16 f;
    f.q[0] = *(const u32x4*)(p);
    f.q[1] = *(const u32x4*)(p + 8);
    return f.v;
}

// B fragment (32x16, bf16): runptr -> 16 contiguous K-values for this lane
static __device__ inline v16bf load_bfrag(const unsigned short* runptr) {
    Frag16 f;
    f.q[0] = *(const u32x4*)(runptr);
    f.q[1] = *(const u32x4*)(runptr + 8);
    return f.v;
}

// ---------------- prep kernels ----------------

__global__ __launch_bounds__(256) void prep_weights(
    const float* __restrict__ Wq, const float* __restrict__ Wk,
    const float* __restrict__ Wv, const float* __restrict__ Wm,
    unsigned short* __restrict__ dst) {
    size_t i = ((size_t)blockIdx.x * blockDim.x + threadIdx.x) * 4;
    int m = (int)(i >> 20);
    size_t off = i & ((1u << 20) - 1);
    const float* src = (m == 0) ? Wq : (m == 1) ? Wk : (m == 2) ? Wv : Wm;
    f32x4 x = *(const f32x4*)(src + off);
    u32x2 pk;
    pk[0] = pk2(x[0], x[1]);
    pk[1] = pk2(x[2], x[3]);
    *(u32x2*)(dst + ((size_t)m << 20) + off) = pk;
}

// Convert q/k/v activations fp32 -> bf16 (12M elements, streaming)
__global__ __launch_bounds__(256) void prep_x(
    const float* __restrict__ Xq, const float* __restrict__ Xk,
    const float* __restrict__ Xv, unsigned short* __restrict__ dst) {
    size_t i = ((size_t)blockIdx.x * blockDim.x + threadIdx.x) * 4;
    int m = (int)(i >> 22);                      // 4M elems per tensor
    size_t off = i & ((1u << 22) - 1);
    const float* src = (m == 0) ? Xq : (m == 1) ? Xk : Xv;
    f32x4 x = *(const f32x4*)(src + off);
    u32x2 pk;
    pk[0] = pk2(x[0], x[1]);
    pk[1] = pk2(x[2], x[3]);
    *(u32x2*)(dst + ((size_t)m << 22) + off) = pk;
}

__global__ __launch_bounds__(32) void prep_scales(
    const float* __restrict__ z_phi, const float* __restrict__ u,
    float* __restrict__ hscale) {
    int h = threadIdx.x;
    if (h < NHEAD) {
        float z = z_phi[0];
        float p = 1.f / (1.f + __expf(-z));
        float uu = fminf(fmaxf(u[h], 1e-6f), 1.f - 1e-6f);
        float g = (z + __logf(uu) - log1pf(-uu)) * 10.f;   // /TEMP, TEMP=0.1
        g = 1.f / (1.f + __expf(-g));
        hscale[h] = (1.f - g) / (1.f - p);
    }
}

// ---------------- projection GEMM: Y = X @ W^T + b ----------------
// Each wave: 32(M) x 64(N) output tile -> 8 wmma / K-step, A frags reused 4x, B 2x.
// blockIdx.z: 0=q -> Qh[bh][s][d], 1=k -> Kh[bh][s][d], 2=v -> Vt[bh][d][s]
__global__ __launch_bounds__(128) void proj_kernel(
    const unsigned short* __restrict__ Xall, const unsigned short* __restrict__ Wbf,
    const float* __restrict__ bq, const float* __restrict__ bk,
    const float* __restrict__ bv,
    unsigned short* __restrict__ Qh, unsigned short* __restrict__ Kh,
    unsigned short* __restrict__ Vt) {
    int lane = threadIdx.x & 31, wv = threadIdx.x >> 5;
    int tile = blockIdx.x * 4 + wv;          // 0..1023
    int mt = tile >> 4;                      // 0..63 : 32-row tile
    int nt = tile & 15;                      // 0..15 : 64-col tile == head
    int b = blockIdx.y, which = blockIdx.z;
    const unsigned short* X = Xall + ((size_t)which << 22);
    const unsigned short* W = Wbf + ((size_t)which << 20);
    const float* bias = (which == 0) ? bq : (which == 1) ? bk : bv;

    int n = lane & 15, hh = lane >> 4;
    const unsigned short* Xb = X + (size_t)b * SEQ * HID;

    v8f acc[2][4];
#pragma unroll
    for (int mi = 0; mi < 2; ++mi)
#pragma unroll
        for (int j = 0; j < 4; ++j)
            acc[mi][j] = (v8f){0.f,0.f,0.f,0.f,0.f,0.f,0.f,0.f};

    for (int kb = 0; kb < HID; kb += 32) {
        v16bf a0 = load_afrag_bf(Xb, HID, mt * 32, kb, lane);
        v16bf a1 = load_afrag_bf(Xb, HID, mt * 32 + 16, kb, lane);
#pragma unroll
        for (int j = 0; j < 4; ++j) {
            int o = nt * 64 + j * 16 + n;
            v16bf bf = load_bfrag(W + (size_t)o * HID + kb + 16 * hh);
            acc[0][j] = wmma_bf16(a0, bf, acc[0][j]);
            acc[1][j] = wmma_bf16(a1, bf, acc[1][j]);
        }
    }
    int head = nt;                            // 64-col tile aligns with a head
    if (which < 2) {
        unsigned short* Y = (which == 0) ? Qh : Kh;
        size_t hb = (size_t)(b * NHEAD + head) * SEQ;
#pragma unroll
        for (int j = 0; j < 4; ++j) {
            int d = j * 16 + n;
            float bias_o = bias[head * 64 + d];
#pragma unroll
            for (int mi = 0; mi < 2; ++mi)
#pragma unroll
                for (int r = 0; r < 8; ++r) {
                    int s = mt * 32 + mi * 16 + r + 8 * hh;
                    Y[(hb + s) * DHEAD + d] = f2bf(acc[mi][j][r] + bias_o);
                }
        }
    } else {
        size_t hb = (size_t)(b * NHEAD + head) * DHEAD;
#pragma unroll
        for (int j = 0; j < 4; ++j) {
            int d = j * 16 + n;
            float bias_o = bias[head * 64 + d];
#pragma unroll
            for (int mi = 0; mi < 2; ++mi)
#pragma unroll
                for (int r = 0; r < 8; ++r) {
                    int s = mt * 32 + mi * 16 + r + 8 * hh;
                    Vt[(hb + d) * SEQ + s] = f2bf(acc[mi][j][r] + bias_o);
                }
        }
    }
}

// ---------------- flash attention per (b,h), 16-query tile per wave ----------------
__global__ __launch_bounds__(128) void attn_kernel(
    const unsigned short* __restrict__ Qh, const unsigned short* __restrict__ Kh,
    const unsigned short* __restrict__ Vt, const unsigned char* __restrict__ mask,
    const float* __restrict__ hscale, unsigned short* __restrict__ Ctx) {
    __shared__ unsigned short lds[4][16][64];   // per-wave P tile (bf16)
    int lane = threadIdx.x & 31, wv = threadIdx.x >> 5;
    int qt = blockIdx.x * 4 + wv;
    int bh = blockIdx.y;
    int b = bh >> 4, head = bh & 15;
    const unsigned short* Qb = Qh + (size_t)bh * SEQ * DHEAD;
    const unsigned short* Kb = Kh + (size_t)bh * SEQ * DHEAD;
    const unsigned short* Vb = Vt + (size_t)bh * DHEAD * SEQ;
    const unsigned char* mb = mask + (size_t)b * SEQ;
    int n = lane & 15, hh = lane >> 4;

    // Q A-fragments for d=0..31 and d=32..63 — resident for whole loop
    v16bf qa0 = load_afrag_bf(Qb, DHEAD, qt * 16, 0, lane);
    v16bf qa1 = load_afrag_bf(Qb, DHEAD, qt * 16, 32, lane);

    v8f acc[4];
#pragma unroll
    for (int j = 0; j < 4; ++j) acc[j] = (v8f){0.f,0.f,0.f,0.f,0.f,0.f,0.f,0.f};
    float mrun[8], lrun[8];
#pragma unroll
    for (int r = 0; r < 8; ++r) { mrun[r] = -3.0e38f; lrun[r] = 0.f; }

    for (int k0 = 0; k0 < SEQ; k0 += 64) {
        // ---- scores for 4 key tiles (64 keys), scaled + masked ----
        v8f sc[4];
#pragma unroll
        for (int t = 0; t < 4; ++t) {
            int sn = k0 + t * 16 + n;               // key row for this lane's column
            v8f c = {0.f, 0.f, 0.f, 0.f, 0.f, 0.f, 0.f, 0.f};
            c = wmma_bf16(qa0, load_bfrag(Kb + (size_t)sn * DHEAD + 16 * hh), c);
            c = wmma_bf16(qa1, load_bfrag(Kb + (size_t)sn * DHEAD + 32 + 16 * hh), c);
            bool mk = mb[sn] != 0;
#pragma unroll
            for (int r = 0; r < 8; ++r) {
                float sv = c[r] * 0.125f;           // 1/sqrt(64)
                c[r] = mk ? -1e9f : sv;
            }
            sc[t] = c;
        }
        // ---- online softmax (rows live in 16-lane halves) ----
        float alpha[8];
#pragma unroll
        for (int r = 0; r < 8; ++r) {
            float v = fmaxf(fmaxf(sc[0][r], sc[1][r]), fmaxf(sc[2][r], sc[3][r]));
            v = fmaxf(v, __shfl_xor(v, 1, 32));
            v = fmaxf(v, __shfl_xor(v, 2, 32));
            v = fmaxf(v, __shfl_xor(v, 4, 32));
            v = fmaxf(v, __shfl_xor(v, 8, 32));
            float mn = fmaxf(mrun[r], v);
            alpha[r] = __expf(mrun[r] - mn);
            mrun[r] = mn;
        }
        float rsum[8];
#pragma unroll
        for (int r = 0; r < 8; ++r) rsum[r] = 0.f;
#pragma unroll
        for (int t = 0; t < 4; ++t) {
#pragma unroll
            for (int r = 0; r < 8; ++r) {
                float p = __expf(sc[t][r] - mrun[r]);
                sc[t][r] = p;
                rsum[r] += p;
            }
        }
#pragma unroll
        for (int r = 0; r < 8; ++r) {
            float v = rsum[r];
            v += __shfl_xor(v, 1, 32);
            v += __shfl_xor(v, 2, 32);
            v += __shfl_xor(v, 4, 32);
            v += __shfl_xor(v, 8, 32);
            lrun[r] = lrun[r] * alpha[r] + v;
        }
#pragma unroll
        for (int r = 0; r < 8; ++r) {
            acc[0][r] *= alpha[r]; acc[1][r] *= alpha[r];
            acc[2][r] *= alpha[r]; acc[3][r] *= alpha[r];
        }
        // ---- stage P (C-layout -> A-layout) through LDS as bf16 ----
#pragma unroll
        for (int t = 0; t < 4; ++t)
#pragma unroll
            for (int r = 0; r < 8; ++r)
                lds[wv][r + 8 * hh][t * 16 + n] = f2bf(sc[t][r]);
        __builtin_amdgcn_wave_barrier();
        // P A-fragments (keys k0..k0+31 and k0+32..k0+63)
        Frag16 p0, p1;
        {
            const unsigned short* lp = &lds[wv][lane & 15][0];
            p0.q[0] = *(const u32x4*)(lp + 16 * hh);
            p0.q[1] = *(const u32x4*)(lp + 8 + 16 * hh);
            p1.q[0] = *(const u32x4*)(lp + 32 + 16 * hh);
            p1.q[1] = *(const u32x4*)(lp + 40 + 16 * hh);
        }
        __builtin_amdgcn_wave_barrier();
        // ---- out += P @ V  (V is transposed [d][s] -> contiguous B runs) ----
#pragma unroll
        for (int j = 0; j < 4; ++j) {
            const unsigned short* vp = Vb + (size_t)(j * 16 + n) * SEQ + k0 + 16 * hh;
            acc[j] = wmma_bf16(p0.v, load_bfrag(vp), acc[j]);
            acc[j] = wmma_bf16(p1.v, load_bfrag(vp + 32), acc[j]);
        }
    }
    // ---- epilogue: normalize, apply per-head concrete-dropout scale, store ctx ----
    float hs = hscale[head];
    float inv[8];
#pragma unroll
    for (int r = 0; r < 8; ++r) inv[r] = hs / lrun[r];
    unsigned short* cp = Ctx + (size_t)b * SEQ * HID + head * DHEAD;
#pragma unroll
    for (int j = 0; j < 4; ++j) {
#pragma unroll
        for (int r = 0; r < 8; ++r) {
            int s = qt * 16 + r + 8 * hh;
            cp[(size_t)s * HID + j * 16 + n] = f2bf(acc[j][r] * inv[r]);
        }
    }
}

// ---------------- output projection: out = ctx @ Wm^T + bm (fp32) ----------------
__global__ __launch_bounds__(128) void outproj_kernel(
    const unsigned short* __restrict__ Ctx, const unsigned short* __restrict__ Wm,
    const float* __restrict__ bm, float* __restrict__ Out) {
    int lane = threadIdx.x & 31, wv = threadIdx.x >> 5;
    int tile = blockIdx.x * 4 + wv;          // 0..1023
    int mt = tile >> 4;                      // 0..63
    int nt = tile & 15;                      // 0..15
    int b = blockIdx.y;
    int n = lane & 15, hh = lane >> 4;
    const unsigned short* A = Ctx + (size_t)b * SEQ * HID;

    v8f acc[2][4];
#pragma unroll
    for (int mi = 0; mi < 2; ++mi)
#pragma unroll
        for (int j = 0; j < 4; ++j)
            acc[mi][j] = (v8f){0.f,0.f,0.f,0.f,0.f,0.f,0.f,0.f};

    for (int kb = 0; kb < HID; kb += 32) {
        v16bf a0 = load_afrag_bf(A, HID, mt * 32, kb, lane);
        v16bf a1 = load_afrag_bf(A, HID, mt * 32 + 16, kb, lane);
#pragma unroll
        for (int j = 0; j < 4; ++j) {
            int o = nt * 64 + j * 16 + n;
            v16bf bf = load_bfrag(Wm + (size_t)o * HID + kb + 16 * hh);
            acc[0][j] = wmma_bf16(a0, bf, acc[0][j]);
            acc[1][j] = wmma_bf16(a1, bf, acc[1][j]);
        }
    }
#pragma unroll
    for (int j = 0; j < 4; ++j) {
        int o = nt * 64 + j * 16 + n;
        float bo = bm[o];
#pragma unroll
        for (int mi = 0; mi < 2; ++mi)
#pragma unroll
            for (int r = 0; r < 8; ++r) {
                int s = mt * 32 + mi * 16 + r + 8 * hh;
                Out[((size_t)(b * SEQ + s)) * HID + o] = acc[mi][j][r] + bo;
            }
    }
}

extern "C" void kernel_launch(void* const* d_in, const int* in_sizes, int n_in,
                              void* d_out, int out_size, void* d_ws, size_t ws_size,
                              hipStream_t stream) {
    (void)in_sizes; (void)n_in; (void)out_size; (void)ws_size;
    const float* v  = (const float*)d_in[0];
    const float* k  = (const float*)d_in[1];
    const float* q  = (const float*)d_in[2];
    const unsigned char* mask = (const unsigned char*)d_in[3];   // bool array
    const float* Wv = (const float*)d_in[4];
    const float* bv = (const float*)d_in[5];
    const float* Wk = (const float*)d_in[6];
    const float* bk = (const float*)d_in[7];
    const float* Wq = (const float*)d_in[8];
    const float* bq = (const float*)d_in[9];
    const float* Wm = (const float*)d_in[10];
    const float* bm = (const float*)d_in[11];
    const float* z_phi = (const float*)d_in[12];
    const float* u  = (const float*)d_in[13];
    float* out = (float*)d_out;

    char* ws = (char*)d_ws;
    size_t tensElems = (size_t)BATCH * SEQ * HID;                   // 4M elems = 8MB bf16
    unsigned short* wbf = (unsigned short*)ws;                      // 4 x 1M bf16 = 8MB
    float* hscale = (float*)(ws + ((size_t)8 << 20));               // 256B slot
    unsigned short* Qh  = (unsigned short*)(ws + ((size_t)8 << 20) + 256);
    unsigned short* Kh  = Qh + tensElems;
    unsigned short* Vt  = Kh + tensElems;
    unsigned short* Xbf = Vt + tensElems;     // Xq,Xk,Xv bf16 (3 x 8MB)
    unsigned short* Ctx = Xbf;                // alias: Xq_bf dead once attn starts

    prep_weights<<<4096, 256, 0, stream>>>(Wq, Wk, Wv, Wm, wbf);
    prep_x<<<12288, 256, 0, stream>>>(q, k, v, Xbf);
    prep_scales<<<1, 32, 0, stream>>>(z_phi, u, hscale);
    proj_kernel<<<dim3(256, BATCH, 3), 128, 0, stream>>>(Xbf, wbf, bq, bk, bv,
                                                         Qh, Kh, Vt);
    attn_kernel<<<dim3(SEQ / 16 / 4, BATCH * NHEAD), 128, 0, stream>>>(Qh, Kh, Vt, mask,
                                                                       hscale, Ctx);
    outproj_kernel<<<dim3(256, BATCH), 128, 0, stream>>>(Ctx, wbf + ((size_t)3 << 20),
                                                         bm, out);
}
